// GPConv3D_23467701305627
// MI455X (gfx1250) — matrix operand order repaired and checked by reading, not compiled, not run on hardware
//
#include <hip/hip_runtime.h>

// ---------------------------------------------------------------------------
// Geometric-algebra 3D conv (Cl(3,0)) as implicit GEMM on gfx1250 WMMA.
// Fused problem: N=8, Cf=64, Of=64, 3x3x3 valid conv, 48^3 -> 46^3, fp32.
// Strategy: bf16x3 split (hi*hi + hi*lo + lo*hi) on v_wmma_f32_16x16x32_bf16,
// f32 accumulation. Compute-bound (405 FLOP/B >> machine balance @23.3TB/s).
// A-fragment (weight) stream covered by WGP-scope global_prefetch_b8 (inline
// asm; the builtin only reaches SE scope) issued 6 K-steps ahead, so the
// scheduler-sunk global_load_b128 consume WGP$-resident lines.
// ---------------------------------------------------------------------------

typedef __attribute__((ext_vector_type(16))) __bf16 v16bf;
typedef __attribute__((ext_vector_type(8)))  __bf16 v8bf;
typedef __attribute__((ext_vector_type(8)))  float  v8f;

#define DIN   48
#define DIN2  (48 * 48)
#define DIN3  (48 * 48 * 48)
#define DOUT  46
#define CPAD  72                    // padded channel stride (bank spread, 16B align)
#define XP    50                    // x extent of staged patch (48 + 2 halo, zero tail)
#define PATCH_ELEMS (9 * XP * CPAD) // 32,400 bf16 per copy (hi / lo)
#define NKT   54                    // K = 27 taps * 64 ch = 1728 = 54 * 32
#define PFDIST 6                    // A-stream prefetch distance in K-steps
#define WFRAG_ELEMS (NKT * 4 * 32 * 16) // 110,592 bf16 per copy

__device__ __forceinline__ float bf2f(__bf16 h) {
  unsigned short s = __builtin_bit_cast(unsigned short, h);
  unsigned u = ((unsigned)s) << 16;
  return __builtin_bit_cast(float, u);
}
__device__ __forceinline__ __bf16 f2bf(float f) {  // round-to-nearest-even
  unsigned u = __builtin_bit_cast(unsigned, f);
  unsigned r = (u + 0x7FFFu + ((u >> 16) & 1u)) >> 16;
  unsigned short s = (unsigned short)r;
  return __builtin_bit_cast(__bf16, s);
}

// WGP-scope prefetch: SCOPE=0 pulls the line into ALL cache levels (WGP$ + GL2).
// __builtin_prefetch only emits scope:SCOPE_SE (GL2-only), so use inline asm.
__device__ __forceinline__ void prefetch_wgp(const void* p) {
  asm volatile("global_prefetch_b8 %0, off" : : "v"(p));
}

// 32B LDS fragment load as two 16B halves (rows are 16B- but not 32B-aligned).
__device__ __forceinline__ v16bf ld16(const __bf16* p) {
  v8bf a = *(const v8bf*)p;
  v8bf b = *(const v8bf*)(p + 8);
  return __builtin_shufflevector(a, b, 0, 1, 2, 3, 4, 5, 6, 7,
                                 8, 9, 10, 11, 12, 13, 14, 15);
}

#define WMMA_BF16(A, B, C) \
  __builtin_amdgcn_wmma_f32_16x16x32_bf16(false, (A), false, (B), (short)0, (C), false, false)

// ---------------------------------------------------------------------------
// Prep: fold G into W (j = i^k gather with sign), split to bf16 hi/lo, and lay
// out directly as per-lane A-matrix (16x32) fragments, contiguous k order:
//   lanes 0-15: V0..7 = K0..15 (pairs)
//   lanes16-31: V0..7 = K16..31
// Fragment stream index t = ((kt*4 + mt)*32 + lane)*16 + slot.
// K ordering: k = tap*64 + cf  (tap = (m*3+l)*3+v), so K-tile kt -> tap=kt>>1,
// channel base (kt&1)*32.
// ---------------------------------------------------------------------------
__global__ void ga_prep_wfrag(const float* __restrict__ W,
                              const float* __restrict__ G,
                              __bf16* __restrict__ WH,
                              __bf16* __restrict__ WL) {
  int t = blockIdx.x * 256 + threadIdx.x;
  if (t >= WFRAG_ELEMS) return;
  int s    = t & 15;
  int lane = (t >> 4) & 31;
  int mt   = (t >> 9) & 3;
  int kt   = t >> 11;

  int vreg = s >> 1, h = s & 1;
  int hi8  = (lane & 16) ? 8 : 0;
  int kk   = 2 * hi8 + 2 * vreg + h;

  int tap = kt >> 1;
  int cf  = ((kt & 1) << 5) + kk;          // fused input channel 0..63
  int oc  = (mt << 4) + (lane & 15);       // fused output channel 0..63

  int m = tap / 9, l = (tap / 3) % 3, v = tap % 3;
  int o = oc >> 3, kb = oc & 7;
  int c = cf >> 3, i = cf & 7;
  int j = i ^ kb;                          // only surviving G term
  float sign = G[(i * 8 + j) * 8 + kb];
  float wv   = W[((((m * 3 + l) * 3 + v) * 8 + c) * 8 + o) * 8 + j];
  float full = sign * wv;
  __bf16 hh = f2bf(full);
  __bf16 ll = f2bf(full - bf2f(hh));
  WH[t] = hh;
  WL[t] = ll;
}

// ---------------------------------------------------------------------------
// Main: one workgroup (4 waves / 128 thr) per (batch, z, y) output row.
// Wave mt owns oc tile [16mt,16mt+16); 3 N-tiles of 16 x-positions (48 cols,
// lanes with x>=46 computed but never stored). 54 K-steps * 9 WMMA each.
// ---------------------------------------------------------------------------
__global__ __launch_bounds__(128)
void ga_conv3d_wmma(const float* __restrict__ x,
                    const float* __restrict__ bias,
                    const __bf16* __restrict__ WH,
                    const __bf16* __restrict__ WL,
                    float* __restrict__ out) {
  extern __shared__ __align__(16) char smem[];
  __bf16* sPh = (__bf16*)smem;
  __bf16* sPl = sPh + PATCH_ELEMS;

  const int tid = threadIdx.x;
  const int z = blockIdx.x, y = blockIdx.y, bb = blockIdx.z;

  // ---- Stage input patch [dz=3][dy=3][x=50][c=64(pad 72)] as bf16 hi/lo ----
  const float* X = x + (size_t)bb * 64 * DIN3;
  for (int idx = tid; idx < 64 * 9 * XP; idx += 128) {
    int xl   = idx % XP;
    int rest = idx / XP;          // cf*9 + dz*3 + dy  (x fastest: coalesced gmem)
    int dzdy = rest % 9;
    int cf   = rest / 9;
    int dz = dzdy / 3, dy = dzdy % 3;
    float val = 0.0f;
    if (xl < DIN)
      val = X[(size_t)cf * DIN3 + (z + dz) * DIN2 + (y + dy) * DIN + xl];
    __bf16 hh = f2bf(val);
    __bf16 ll = f2bf(val - bf2f(hh));
    int la = ((dz * 3 + dy) * XP + xl) * CPAD + cf;
    sPh[la] = hh;
    sPl[la] = ll;
  }
  __syncthreads();

  const int lane  = tid & 31;
  const int mt    = tid >> 5;
  const int n     = lane & 15;        // N column (x within tile)
  const int khalf = (lane >> 4) & 1;  // B: lanes 16-31 carry K 16..31

  v8f acc0 = {0.f, 0.f, 0.f, 0.f, 0.f, 0.f, 0.f, 0.f};
  v8f acc1 = acc0, acc2 = acc0;

  const __bf16* whp = WH + ((size_t)mt * 32 + lane) * 16;
  const __bf16* wlp = WL + ((size_t)mt * 32 + lane) * 16;

  // Warm the A-fragment stream: each lane prefetches its 32B slice, so a wave
  // covers the whole 1KB (kt,mt) block at WGP scope (all cache levels).
#pragma unroll
  for (int p = 0; p < PFDIST; ++p) {
    prefetch_wgp(whp + (size_t)p * 2048);
    prefetch_wgp(wlp + (size_t)p * 2048);
  }

  for (int kt = 0; kt < NKT; ++kt) {
    // Keep the prefetch front PFDIST K-steps (~54 WMMAs) ahead of the loads.
    int ktp = kt + PFDIST;
    if (ktp >= NKT) ktp = NKT - 1;
    prefetch_wgp(whp + (size_t)ktp * 2048);
    prefetch_wgp(wlp + (size_t)ktp * 2048);

    // A fragments: coalesced 32B/lane; WGP$-resident after prefetch, so it is
    // harmless if the scheduler sinks these next to their consuming WMMA.
    v16bf Ah = *(const v16bf*)(whp + (size_t)kt * 2048);
    v16bf Al = *(const v16bf*)(wlp + (size_t)kt * 2048);

    int tap = kt >> 1;
    int m = tap / 9, l = (tap / 3) % 3, v = tap % 3;
    int cbase = ((kt & 1) << 5) + (khalf << 4);
    int e0 = ((m * 3 + l) * XP + n + v) * CPAD + cbase;

    // B fragments (32x16 bf16): 16 contiguous channels per lane, 2x ds_load_b128.
    v16bf Bh0 = ld16(sPh + e0);
    v16bf Bl0 = ld16(sPl + e0);
    v16bf Bh1 = ld16(sPh + e0 + 16 * CPAD);
    v16bf Bl1 = ld16(sPl + e0 + 16 * CPAD);
    v16bf Bh2 = ld16(sPh + e0 + 32 * CPAD);
    v16bf Bl2 = ld16(sPl + e0 + 32 * CPAD);

    // bf16x3: hi*hi + hi*lo + lo*hi, interleaved across 3 independent chains.
    acc0 = WMMA_BF16(Ah, Bh0, acc0);
    acc1 = WMMA_BF16(Ah, Bh1, acc1);
    acc2 = WMMA_BF16(Ah, Bh2, acc2);
    acc0 = WMMA_BF16(Ah, Bl0, acc0);
    acc1 = WMMA_BF16(Ah, Bl1, acc1);
    acc2 = WMMA_BF16(Ah, Bl2, acc2);
    acc0 = WMMA_BF16(Al, Bh0, acc0);
    acc1 = WMMA_BF16(Al, Bh1, acc1);
    acc2 = WMMA_BF16(Al, Bh2, acc2);
  }

  // ---- Epilogue: C/D layout row M = r + 8*(lane>=16), col N = lane&15 ----
  float bs[8];
#pragma unroll
  for (int r = 0; r < 8; ++r) bs[r] = bias[mt * 16 + khalf * 8 + r];

#pragma unroll
  for (int t = 0; t < 3; ++t) {
    int xo = t * 16 + n;
    if (xo < DOUT) {
      v8f a = (t == 0) ? acc0 : ((t == 1) ? acc1 : acc2);
#pragma unroll
      for (int r = 0; r < 8; ++r) {
        int oc = mt * 16 + khalf * 8 + r;
        out[((((size_t)bb * 64 + oc) * DOUT + z) * DOUT + y) * DOUT + xo] =
            a[r] + bs[r];
      }
    }
  }
}

// ---------------------------------------------------------------------------
extern "C" void kernel_launch(void* const* d_in, const int* in_sizes, int n_in,
                              void* d_out, int out_size, void* d_ws, size_t ws_size,
                              hipStream_t stream) {
  const float* x = (const float*)d_in[0];
  const float* W = (const float*)d_in[1];
  const float* b = (const float*)d_in[2];
  const float* G = (const float*)d_in[3];
  float* out = (float*)d_out;

  __bf16* WH = (__bf16*)d_ws;            // needs 2 * 110,592 * 2B = 442,368 B
  __bf16* WL = WH + WFRAG_ELEMS;

  ga_prep_wfrag<<<(WFRAG_ELEMS + 255) / 256, 256, 0, stream>>>(W, G, WH, WL);

  dim3 grid(DOUT, DOUT, 8);              // (z, y, batch)
  size_t shmem = 2 * (size_t)PATCH_ELEMS * sizeof(__bf16);  // 129,600 B
  ga_conv3d_wmma<<<grid, 128, shmem, stream>>>(x, b, WH, WL, out);
}
